// RSAMAttn_67980742361808
// MI455X (gfx1250) — matrix operand order repaired and checked by reading, not compiled, not run on hardware
//
#include <hip/hip_runtime.h>

typedef __attribute__((ext_vector_type(16))) _Float16 v16h;
typedef __attribute__((ext_vector_type(8)))  float    v8f;
typedef __attribute__((ext_vector_type(8)))  _Float16 h8;

#define B_   4
#define H_   128
#define W_   128
#define C_   256
#define N_   (H_*W_)      /* 16384 tokens per batch */
#define WS_  8
#define HD_  32
#define GH_  4
#define LH_  4
#define GD_  128
#define LD_  128
#define M_   256          /* pooled tokens per batch (16x16) */
#define SCALE_ 0.17677669529663687f  /* 32^-0.5 */

// ---------------------------------------------------------------------------
// fp32 -> f16 convert (grid-stride)
// ---------------------------------------------------------------------------
__global__ void k_cvt(const float* __restrict__ in, _Float16* __restrict__ out, int n) {
  int i = blockIdx.x * blockDim.x + threadIdx.x;
  int stride = gridDim.x * blockDim.x;
  for (; i < n; i += stride) out[i] = (_Float16)in[i];
}

// ---------------------------------------------------------------------------
// Weight swizzle (fused fp32->f16): pack W[K,N] into per-lane-contiguous WMMA
// B fragments. Fragment (kc, tn): lane needs B[k = kc*32 + 16*(lane/16) + 2j
// (+1)][n = tn*16 + lane%16] at Wsw[((kc*ntiles+tn)*32+lane)*16 + 2j (+1)].
// ---------------------------------------------------------------------------
__global__ void k_wswz(const float* __restrict__ W, _Float16* __restrict__ Wsw,
                       int N, int K) {
  int idx = blockIdx.x * blockDim.x + threadIdx.x;
  int ntiles = N >> 4;
  int total = (K >> 5) * ntiles * 32 * 8;
  if (idx >= total) return;
  int j    = idx & 7;
  int lane = (idx >> 3) & 31;
  int tile = idx >> 8;
  int tn = tile % ntiles;
  int kc = tile / ntiles;
  int g = lane >> 4;
  int col = tn * 16 + (lane & 15);
  int r = kc * 32 + 16 * g + 2 * j;
  size_t dst = ((size_t)tile * 32 + lane) * 16 + 2 * j;
  Wsw[dst]     = (_Float16)W[(size_t)r * N + col];
  Wsw[dst + 1] = (_Float16)W[(size_t)(r + 1) * N + col];
}

// ---------------------------------------------------------------------------
// V fragment swizzle for global attention AV: per (b, head, c8, tn) fragment,
// lane needs v[key = c8*32 + 16*(lane/16) + 2j (+1)][d = tn*16 + lane%16].
// ---------------------------------------------------------------------------
__global__ void k_vswz(const _Float16* __restrict__ kvg, _Float16* __restrict__ vsw) {
  int idx = blockIdx.x * blockDim.x + threadIdx.x;  // B*GH*8*2*32*8 = 65536
  int j    = idx & 7;
  int lane = (idx >> 3) & 31;
  int tn   = (idx >> 8) & 1;
  int c8   = (idx >> 9) & 7;
  int head = (idx >> 12) & 3;
  int b    = idx >> 14;
  int g = lane >> 4;
  int d = tn * 16 + (lane & 15);
  int key = c8 * 32 + 16 * g + 2 * j;
  const _Float16* vrow = kvg + ((size_t)b * M_ + key) * C_ + GD_ + head * HD_;
  size_t dst = (((size_t)((b * GH_ + head) * 8 + c8) * 2 + tn) * 32 + lane) * 16 + 2 * j;
  vsw[dst]     = vrow[d];
  vsw[dst + 1] = vrow[C_ + d];   // key+1 row
}

// ---------------------------------------------------------------------------
// 8x8 mean pool of x_reverse (fp32 in) -> pooled f16 (B*256, C)
// ---------------------------------------------------------------------------
__global__ void k_pool(const float* __restrict__ xr, _Float16* __restrict__ pooled) {
  int tok = blockIdx.x;              // b*256 + t
  int b = tok >> 8;
  int t = tok & 255;
  int r0 = (t >> 4) * WS_;
  int c0 = (t & 15) * WS_;
  int ch = threadIdx.x;
  float s = 0.f;
  #pragma unroll
  for (int i = 0; i < WS_; ++i)
    #pragma unroll
    for (int j = 0; j < WS_; ++j)
      s += xr[(((size_t)b * H_ + (r0 + i)) * W_ + (c0 + j)) * C_ + ch];
  pooled[(size_t)tok * C_ + ch] = (_Float16)(s * (1.f / 64.f));
}

// ---------------------------------------------------------------------------
// f16 WMMA GEMM on pre-swizzled weights: out = A[M,K] @ B[K,N] (+bias).
// Wave computes a 16x128 tile; all fragment loads are contiguous b128s.
// ---------------------------------------------------------------------------
__global__ __launch_bounds__(256) void k_gemm(
    const _Float16* __restrict__ A, const _Float16* __restrict__ Bsw,
    int M, int N, int K, int nchunks, const float* __restrict__ bias,
    _Float16* __restrict__ out16, float* __restrict__ out32,
    int ostride, int ocol) {
  int wave = threadIdx.x >> 5;
  int lane = threadIdx.x & 31;
  int gidx = blockIdx.x * (blockDim.x >> 5) + wave;
  int work = (M >> 4) * nchunks;
  if (gidx >= work) return;                 // wave-uniform
  int tile_m = gidx / nchunks;
  int nc = gidx - tile_m * nchunks;
  int m = lane & 15;
  int g = lane >> 4;
  int ntiles = N >> 4;

  v8f acc[8] = {};
  const _Float16* arow = A + (size_t)(tile_m * 16 + m) * K + 16 * g;
  int kchunks = K >> 5;
  for (int kc = 0; kc < kchunks; ++kc) {
    __builtin_prefetch(arow + (kc + 1) * 32, 0, 1);
    h8 alo = *(const h8*)(arow + kc * 32);
    h8 ahi = *(const h8*)(arow + kc * 32 + 8);
    v16h a;
    #pragma unroll
    for (int e = 0; e < 8; ++e) { a[e] = alo[e]; a[8 + e] = ahi[e]; }
    const _Float16* bfrag = Bsw + (((size_t)kc * ntiles + nc * 8) * 32 + lane) * 16;
    #pragma unroll
    for (int t = 0; t < 8; ++t) {
      v16h bm = *(const v16h*)(bfrag + t * 512);
      acc[t] = __builtin_amdgcn_wmma_f32_16x16x32_f16(
          false, a, false, bm, (short)0, acc[t], false, false);
    }
  }
  int orow_base = tile_m * 16 + 8 * g;      // C: M = vgpr + 8*(lane/16)
  #pragma unroll
  for (int t = 0; t < 8; ++t) {
    int col = nc * 128 + t * 16 + m;
    #pragma unroll
    for (int v = 0; v < 8; ++v) {
      int orow = orow_base + v;
      float val = acc[t][v];
      if (out32) out32[(size_t)orow * ostride + ocol + col] = val + (bias ? bias[col] : 0.f);
      else       out16[(size_t)orow * N + col] = (_Float16)val;
    }
  }
}

// ---------------------------------------------------------------------------
// Global attention: per block (1 wave) one (b, head, 16-query tile).
// ---------------------------------------------------------------------------
#define SSTR 257   /* padded fp32 score row stride */
#define PSTR 264   /* padded f16 prob row stride (16B-aligned fragments) */

__global__ __launch_bounds__(32) void k_gattn(
    const _Float16* __restrict__ q, const _Float16* __restrict__ kvg,
    const _Float16* __restrict__ vsw, _Float16* __restrict__ gout) {
  __shared__ float    sS[16 * SSTR];
  __shared__ _Float16 sP[16 * PSTR];

  int qtile = blockIdx.x & 1023;
  int head  = (blockIdx.x >> 10) & 3;
  int b     = blockIdx.x >> 12;
  int lane  = threadIdx.x;
  int m = lane & 15;
  int g = lane >> 4;

  // pack A = q tile (16 x 32): two contiguous 16B chunks per lane
  const _Float16* arow = q + ((size_t)b * N_ + qtile * 16 + m) * GD_ + head * HD_ + 16 * g;
  h8 qlo = *(const h8*)(arow);
  h8 qhi = *(const h8*)(arow + 8);
  v16h a;
  #pragma unroll
  for (int e = 0; e < 8; ++e) { a[e] = qlo[e]; a[8 + e] = qhi[e]; }

  const _Float16* kbase = kvg + (size_t)b * M_ * C_ + head * HD_;

  // scores: 16 key tiles of 16; B fragment is 16 contiguous halves of k row
  for (int t = 0; t < 16; ++t) {
    const _Float16* krow = kbase + (size_t)(t * 16 + m) * C_ + 16 * g;
    v16h bm = *(const v16h*)krow;
    v8f c = {};
    c = __builtin_amdgcn_wmma_f32_16x16x32_f16(
        false, a, false, bm, (short)0, c, false, false);
    #pragma unroll
    for (int v = 0; v < 8; ++v)
      sS[(v + 8 * g) * SSTR + t * 16 + m] = c[v];
  }
  __syncthreads();

  // softmax: each lane handles half of one row; combine halves via shfl
  {
    int base = m * SSTR + g * 128;
    float mx = -1e30f;
    for (int i = 0; i < 128; ++i) {
      float s = sS[base + i] * SCALE_;
      sS[base + i] = s;
      mx = fmaxf(mx, s);
    }
    mx = fmaxf(mx, __shfl_xor(mx, 16, 32));
    float sum = 0.f;
    for (int i = 0; i < 128; ++i) {
      float e = __expf(sS[base + i] - mx);
      sS[base + i] = e;
      sum += e;
    }
    sum += __shfl_xor(sum, 16, 32);
    float inv = 1.f / sum;
    int pbase = m * PSTR + g * 128;
    for (int i = 0; i < 128; ++i)
      sP[pbase + i] = (_Float16)(sS[base + i] * inv);
  }
  __syncthreads();

  // AV: out(16x32) = P(16x256) @ V(256x32); V fragments pre-swizzled
  const _Float16* vfrag = vsw + (size_t)((b * GH_ + head) * 8) * 2 * 32 * 16;
  for (int tn = 0; tn < 2; ++tn) {
    v8f c = {};
    #pragma unroll
    for (int c8 = 0; c8 < 8; ++c8) {
      const _Float16* prow = sP + m * PSTR + c8 * 32 + 16 * g;
      h8 plo = *(const h8*)(prow);
      h8 phi = *(const h8*)(prow + 8);
      v16h pa;
      #pragma unroll
      for (int e = 0; e < 8; ++e) { pa[e] = plo[e]; pa[8 + e] = phi[e]; }
      v16h vb = *(const v16h*)(vfrag + ((size_t)(c8 * 2 + tn) * 32 + lane) * 16);
      c = __builtin_amdgcn_wmma_f32_16x16x32_f16(
          false, pa, false, vb, (short)0, c, false, false);
    }
    #pragma unroll
    for (int v = 0; v < 8; ++v)
      gout[((size_t)b * N_ + qtile * 16 + v + 8 * g) * GD_ + head * HD_ + tn * 16 + m] =
          (_Float16)c[v];
  }
}

// ---------------------------------------------------------------------------
// Local branch: per (pixel, head) 4x4 cross-head attention, scatter into the
// reference's scrambled (h,n,d)->(s1,s2,c) layout feeding the Wp_l GEMM.
// ---------------------------------------------------------------------------
__global__ __launch_bounds__(256) void k_lattn(
    const _Float16* __restrict__ ql, const _Float16* __restrict__ kvl,
    _Float16* __restrict__ lpre) {
  int idx = blockIdx.x * blockDim.x + threadIdx.x;   // B*N*LH = 262144
  int h = idx & 3;
  int p = (idx >> 2) & (N_ - 1);
  int b = idx >> 16;

  const h8* q8 = (const h8*)(ql  + ((size_t)b * N_ + p) * LD_ + h * HD_);
  const h8* k8 = (const h8*)(kvl + ((size_t)b * N_ + p) * (2 * LD_));  // k: 0..15, v: 16..31

  h8 qv[4];
  #pragma unroll
  for (int c = 0; c < 4; ++c) qv[c] = q8[c];

  float s[4];
  #pragma unroll
  for (int k = 0; k < 4; ++k) {
    float acc = 0.f;
    #pragma unroll
    for (int c = 0; c < 4; ++c) {
      h8 kk = k8[k * 4 + c];
      #pragma unroll
      for (int e = 0; e < 8; ++e) acc += (float)qv[c][e] * (float)kk[e];
    }
    s[k] = acc * SCALE_;
  }
  float mx = fmaxf(fmaxf(s[0], s[1]), fmaxf(s[2], s[3]));
  float e0 = __expf(s[0] - mx), e1 = __expf(s[1] - mx);
  float e2 = __expf(s[2] - mx), e3 = __expf(s[3] - mx);
  float inv = 1.f / (e0 + e1 + e2 + e3);
  float w[4] = {e0 * inv, e1 * inv, e2 * inv, e3 * inv};

  float o[HD_];
  #pragma unroll
  for (int d = 0; d < HD_; ++d) o[d] = 0.f;
  #pragma unroll
  for (int k = 0; k < 4; ++k) {
    #pragma unroll
    for (int c = 0; c < 4; ++c) {
      h8 vv = k8[16 + k * 4 + c];
      #pragma unroll
      for (int e = 0; e < 8; ++e) o[c * 8 + e] += w[k] * (float)vv[e];
    }
  }

  // scatter: pixel p=(r,cc), window token n=i*8+j
  int r  = p >> 7, cc = p & 127;
  int gh = r >> 3, i = r & 7, gw = cc >> 3, j = cc & 7;
  int n  = i * 8 + j;
  int s1 = h * 2 + (n >> 5);
  int s2 = (n >> 2) & 7;
  int outp  = (gh * 8 + s1) * W_ + gw * 8 + s2;
  int cbase = (n & 3) * HD_;
  h8* o8 = (h8*)(lpre + ((size_t)b * N_ + outp) * LD_ + cbase);
  #pragma unroll
  for (int c = 0; c < 4; ++c) {
    h8 t;
    #pragma unroll
    for (int e = 0; e < 8; ++e) t[e] = (_Float16)o[c * 8 + e];
    o8[c] = t;
  }
}

// ---------------------------------------------------------------------------
extern "C" void kernel_launch(void* const* d_in, const int* in_sizes, int n_in,
                              void* d_out, int out_size, void* d_ws, size_t ws_size,
                              hipStream_t stream) {
  (void)in_sizes; (void)n_in; (void)out_size; (void)ws_size;
  const float* x     = (const float*)d_in[0];
  const float* xr    = (const float*)d_in[1];
  const float* Wq_g  = (const float*)d_in[2];
  const float* Wkv_g = (const float*)d_in[3];
  const float* Wp_g  = (const float*)d_in[4];
  const float* bp_g  = (const float*)d_in[5];
  const float* Wq_l  = (const float*)d_in[6];
  const float* Wkv_l = (const float*)d_in[7];
  const float* Wp_l  = (const float*)d_in[8];
  const float* bp_l  = (const float*)d_in[9];
  float* out = (float*)d_out;

  char* ws = (char*)d_ws;
  size_t off = 0;
  auto alloc = [&](size_t bytes) -> char* {
    char* p = ws + off;
    off += (bytes + 255) & ~(size_t)255;
    return p;
  };

  _Float16* x16    = (_Float16*)alloc((size_t)B_ * N_ * C_ * 2);
  _Float16* xr16   = (_Float16*)alloc((size_t)B_ * N_ * C_ * 2);
  _Float16* wqg16  = (_Float16*)alloc((size_t)C_ * GD_ * 2);
  _Float16* wkvg16 = (_Float16*)alloc((size_t)C_ * 2 * GD_ * 2);
  _Float16* wpg16  = (_Float16*)alloc((size_t)GD_ * GD_ * 2);
  _Float16* wql16  = (_Float16*)alloc((size_t)C_ * LD_ * 2);
  _Float16* wkvl16 = (_Float16*)alloc((size_t)C_ * 2 * LD_ * 2);
  _Float16* wpl16  = (_Float16*)alloc((size_t)LD_ * LD_ * 2);
  _Float16* pool16 = (_Float16*)alloc((size_t)B_ * M_ * C_ * 2);
  _Float16* qg16   = (_Float16*)alloc((size_t)B_ * N_ * GD_ * 2);
  _Float16* kvg16  = (_Float16*)alloc((size_t)B_ * M_ * C_ * 2);
  _Float16* vsw16  = (_Float16*)alloc((size_t)B_ * GH_ * 8 * 2 * 32 * 16 * 2);
  _Float16* ql16   = (_Float16*)alloc((size_t)B_ * N_ * LD_ * 2);
  _Float16* kvl16  = (_Float16*)alloc((size_t)B_ * N_ * 2 * LD_ * 2);
  _Float16* g16    = (_Float16*)alloc((size_t)B_ * N_ * GD_ * 2);
  _Float16* lpre16 = (_Float16*)alloc((size_t)B_ * N_ * LD_ * 2);

  // input conversions + weight swizzles (fused convert)
  k_cvt<<<8192, 256, 0, stream>>>(x,  x16,  B_ * N_ * C_);
  k_cvt<<<8192, 256, 0, stream>>>(xr, xr16, B_ * N_ * C_);
  auto wswz = [&](const float* W, _Float16* Wsw, int N, int K) {
    int total = (K >> 5) * (N >> 4) * 32 * 8;
    k_wswz<<<(total + 255) / 256, 256, 0, stream>>>(W, Wsw, N, K);
  };
  wswz(Wq_g,  wqg16,  GD_,     C_);
  wswz(Wkv_g, wkvg16, 2 * GD_, C_);
  wswz(Wp_g,  wpg16,  GD_,     GD_);
  wswz(Wq_l,  wql16,  LD_,     C_);
  wswz(Wkv_l, wkvl16, 2 * LD_, C_);
  wswz(Wp_l,  wpl16,  LD_,     LD_);
  k_pool<<<B_ * M_, C_, 0, stream>>>(xr, pool16);

  auto gemm = [&](const _Float16* A, const _Float16* Bsw, int M, int N, int K,
                  const float* bias, _Float16* o16, float* o32, int ostr, int ocol) {
    int nchunks = N / 128;
    int work = (M / 16) * nchunks;
    int blocks = (work + 7) / 8;
    k_gemm<<<blocks, 256, 0, stream>>>(A, Bsw, M, N, K, nchunks, bias, o16, o32, ostr, ocol);
  };

  // QKV projections
  gemm(x16,    wqg16,  B_ * N_, GD_,      C_, nullptr, qg16,  nullptr, 0, 0);
  gemm(x16,    wql16,  B_ * N_, LD_,      C_, nullptr, ql16,  nullptr, 0, 0);
  gemm(xr16,   wkvl16, B_ * N_, 2 * LD_,  C_, nullptr, kvl16, nullptr, 0, 0);
  gemm(pool16, wkvg16, B_ * M_, 2 * GD_,  C_, nullptr, kvg16, nullptr, 0, 0);

  // V fragment swizzle for AV pass
  k_vswz<<<(B_ * GH_ * 8 * 2 * 32 * 8) / 256, 256, 0, stream>>>(kvg16, vsw16);

  // attention
  k_gattn<<<B_ * GH_ * (N_ / 16), 32, 0, stream>>>(qg16, kvg16, vsw16, g16);
  k_lattn<<<(B_ * N_ * LH_) / 256, 256, 0, stream>>>(ql16, kvl16, lpre16);

  // output projections, concatenated channels: [0:128)=global, [128:256)=local
  gemm(g16,    wpg16, B_ * N_, GD_, GD_, bp_g, nullptr, out, C_, 0);
  gemm(lpre16, wpl16, B_ * N_, LD_, LD_, bp_l, nullptr, out, C_, 128);
}
